// VAE_28561532518547
// MI455X (gfx1250) — matrix operand order repaired and checked by reading, not compiled, not run on hardware
//
#include <hip/hip_runtime.h>
#include <hip/hip_bf16.h>

// Problem sizes (fixed by the reference)
#define TT     256
#define BBATCH 256
#define HH     512
#define G4H    2048
#define VV     28

typedef __attribute__((ext_vector_type(16))) __bf16 bf16x16;
typedef __attribute__((ext_vector_type(8)))  __bf16 bf16x8;
typedef __attribute__((ext_vector_type(8)))  float  f32x8;
typedef __attribute__((__vector_size__(16))) int    i32x4;

// ---- async global->LDS (CDNA5) with compile-safe fallback ----------------
#if defined(__has_builtin)
#if __has_builtin(__builtin_amdgcn_global_load_async_to_lds_b128)
#define HAVE_ASYNC_LDS 1
#endif
#if __has_builtin(__builtin_amdgcn_s_wait_asynccnt)
#define HAVE_WAIT_ASYNC 1
#endif
#endif
#ifndef HAVE_ASYNC_LDS
#define HAVE_ASYNC_LDS 0
#endif
#ifndef HAVE_WAIT_ASYNC
#define HAVE_WAIT_ASYNC 0
#endif

#if HAVE_ASYNC_LDS
typedef __attribute__((address_space(1))) i32x4 gas_i32x4;  // global
typedef __attribute__((address_space(3))) i32x4 las_i32x4;  // LDS
#endif

__device__ __forceinline__ void copy32_g2lds(const __bf16* g, __bf16* l) {
#if HAVE_ASYNC_LDS
  __builtin_amdgcn_global_load_async_to_lds_b128(
      (gas_i32x4*)g, (las_i32x4*)l, 0, 0);
  __builtin_amdgcn_global_load_async_to_lds_b128(
      (gas_i32x4*)(g + 8), (las_i32x4*)(l + 8), 0, 0);
#else
  *(bf16x16*)l = *(const bf16x16*)g;  // VGPR bounce fallback
#endif
}

__device__ __forceinline__ void async_stage_wait() {
#if HAVE_ASYNC_LDS
#if HAVE_WAIT_ASYNC
  __builtin_amdgcn_s_wait_asynccnt(0);
#else
  asm volatile("s_wait_asynccnt 0" ::: "memory");
#endif
#endif
}

__device__ __forceinline__ __bf16 f2bf(float f) {
  union { float f; unsigned u; } v; v.f = f;
  unsigned r = (v.u + 0x7FFFu + ((v.u >> 16) & 1u)) >> 16;  // RNE
  unsigned short s = (unsigned short)r;
  __bf16 b;
  __builtin_memcpy(&b, &s, sizeof(b));
  return b;
}

// A-fragment (16x32 bf16, row-major source): lane l<16 holds row M=l with
// K = {kb+0..7, kb+16..23}; lane l+16 holds K = {kb+8..15, kb+24..31}.
// Caller passes p = &A[row*ldA + kb + 8*(lane>>4)].
__device__ __forceinline__ bf16x16 ld_afrag(const __bf16* p) {
  bf16x8 lo = *(const bf16x8*)p;
  bf16x8 hi = *(const bf16x8*)(p + 16);
  return __builtin_shufflevector(lo, hi, 0,1,2,3,4,5,6,7,8,9,10,11,12,13,14,15);
}

// B-fragment: 32 lanes x 16 bf16 contiguous per tile (global packed or LDS).
__device__ __forceinline__ bf16x16 ld_bfrag(const __bf16* p) {
  bf16x8 lo = *(const bf16x8*)p;
  bf16x8 hi = *(const bf16x8*)(p + 8);
  return __builtin_shufflevector(lo, hi, 0,1,2,3,4,5,6,7,8,9,10,11,12,13,14,15);
}

__device__ __forceinline__ f32x8 wmma_bf16(bf16x16 a, bf16x16 b, f32x8 c) {
  return __builtin_amdgcn_wmma_f32_16x16x32_bf16(false, a, false, b, (short)0, c,
                                                 false, false);
}

__device__ __forceinline__ float sigm(float x) { return 1.f / (1.f + __expf(-x)); }

// -------------------------------------------------------------------------
// Pack row-major f32 weight [K, N] into WMMA B-fragment order (bf16), with
// column padding to Npad. Tile (kt, nt) stored at ((nt*KT + kt)*32 + lane)*16.
// -------------------------------------------------------------------------
__global__ void pack_b_kernel(const float* __restrict__ src, __bf16* __restrict__ dst,
                              int K, int N, int Npad) {
  int idx = blockIdx.x * blockDim.x + threadIdx.x;
  int total = Npad * K;
  if (idx >= total) return;
  int e    = idx & 15;
  int lane = (idx >> 4) & 31;
  int tile = idx >> 9;
  int KT   = K >> 5;
  int kt   = tile % KT;
  int nt   = tile / KT;
  int lm   = lane & 15, lh = lane >> 4;
  int k = (kt << 5) + (lh << 3) + (e & 7) + ((e >> 3) << 4);
  int n = (nt << 4) + lm;
  float v = (n < N) ? src[(size_t)k * N + n] : 0.f;
  dst[idx] = f2bf(v);
}

// proj[v][n] = sum_k embed[v][k] * Wi[k][n] + b[n]   (tiny: 28 x 2048, full f32)
__global__ void embed_proj_kernel(const float* __restrict__ embed,
                                  const float* __restrict__ Wi,
                                  const float* __restrict__ b,
                                  float* __restrict__ proj) {
  int idx = blockIdx.x * blockDim.x + threadIdx.x;
  if (idx >= VV * G4H) return;
  int v = idx >> 11;
  int n = idx & (G4H - 1);
  float acc = b[n];
  const float* e = embed + (size_t)v * HH;
  for (int k = 0; k < HH; ++k) acc += e[k] * Wi[(size_t)k * G4H + n];
  proj[idx] = acc;
}

__global__ void init_enc_kernel(const float* __restrict__ hidden,
                                __bf16* __restrict__ h0, float* __restrict__ c0) {
  int i = blockIdx.x * blockDim.x + threadIdx.x;
  if (i < BBATCH * HH) { float v = hidden[i]; h0[i] = f2bf(v); c0[i] = v; }
}

__global__ void reparam_kernel(const float* __restrict__ mu, const float* __restrict__ lv,
                               const float* __restrict__ eps,
                               __bf16* __restrict__ z_bf, float* __restrict__ c0) {
  int i = blockIdx.x * blockDim.x + threadIdx.x;
  if (i < BBATCH * HH) {
    float z = mu[i] + eps[i] * __expf(0.5f * lv[i]);
    z_bf[i] = f2bf(z);
    c0[i] = z;
  }
}

// -------------------------------------------------------------------------
// One LSTM timestep, fully fused.  g = proj[token] + h_in @ Wh.
// Block = 128 threads (4 waves along M, M=128 per block), grid (HH/16, B/128).
// B fragments (all 4 gates, 2 K-tiles = 8KB) are async-staged into LDS per
// barrier interval, double-buffered; A fragments pipelined in register banks
// (full unroll -> no cross-stage register moves, no guard branches).
// -------------------------------------------------------------------------
__global__ __launch_bounds__(128) void lstm_step_kernel(
    const __bf16* __restrict__ h_in,   // [B, H] bf16 row-major
    __bf16*       __restrict__ h_out,  // [B, H]
    float*        __restrict__ c_io,   // [B, H] f32 (in-place)
    const __bf16* __restrict__ Whp,    // packed [H, 4H] bf16 fragments
    const float*  __restrict__ proj,   // [V, 4H] f32  (embed@Wi + b)
    const int*    __restrict__ tokens) // [B] or nullptr (=> SOS token 0)
{
  __shared__ __bf16 sB[2][2][4 * 512]; // [buf][k-tile in pair][gate*512+lane*16]
  __shared__ float  sProj[VV * 64];    // proj slice: [tok][gate][16 cols]
  __shared__ int    sTok[128];

  const int tid    = threadIdx.x;
  const int lane   = tid & 31;
  const int wave   = tid >> 5;
  const int lm     = lane & 15;
  const int lh     = lane >> 4;
  const int nhTile = blockIdx.x;            // hidden 16-col tile, 0..31
  const int nh     = nhTile << 4;
  const int mrow0  = blockIdx.y * 128;
  const int mb     = mrow0 + wave * 32;
  const int KT     = HH >> 5;               // 16 K-tiles -> 8 pairs

  // ---- stage tokens + proj slice (consumed after first barrier) ----
  sTok[tid] = tokens ? tokens[mrow0 + tid] : 0;
  for (int i = tid; i < VV * 64; i += 128) {
    int tok = i >> 6, c = i & 63, g = c >> 4, col = c & 15;
    sProj[i] = proj[(size_t)tok * G4H + g * 512 + nh + col];
  }

  // Each thread stages one lane-fragment (32B) of gate `wave` per K-tile.
  const __bf16* gsrc = Whp + (((size_t)(wave * 32 + nhTile) * KT) * 32 + lane) * 16;
  __bf16* const sme0 = &sB[0][0][wave * 512 + lane * 16];
  __bf16* const sme1 = &sB[1][0][wave * 512 + lane * 16];

  // stage pair 0 (K-tiles 0,1) into buffer 0
  copy32_g2lds(gsrc, sme0);
  copy32_g2lds(gsrc + 512, sme0 + 2048);

  // preload previous cell state (consumed only in epilogue)
  float cold[2][8];
  #pragma unroll
  for (int mi = 0; mi < 2; ++mi)
    #pragma unroll
    for (int r = 0; r < 8; ++r)
      cold[mi][r] = c_io[(mrow0 + wave * 32 + mi * 16 + lh * 8 + r) * HH + nh + lm];

  f32x8 acc[2][4] = {};
  bf16x16 aP[2][2][2];                      // [pair parity][k-tile][m-tile]
  #pragma unroll
  for (int t = 0; t < 2; ++t)
    #pragma unroll
    for (int mi = 0; mi < 2; ++mi)
      aP[0][t][mi] = ld_afrag(h_in + (size_t)(mb + mi * 16 + lm) * HH + t * 32 + lh * 8);

  async_stage_wait();
  __syncthreads();

  #pragma unroll
  for (int p = 0; p < 8; ++p) {             // 8 pairs of K-tiles, fully unrolled
    const int cb = p & 1;
    if (p < 7) {
      // stage next pair into the other buffer; pipeline next A fragments
      __bf16* nd = cb ? sme0 : sme1;
      copy32_g2lds(gsrc + (size_t)(2 * p + 2) * 512, nd);
      copy32_g2lds(gsrc + (size_t)(2 * p + 3) * 512, nd + 2048);
      #pragma unroll
      for (int t = 0; t < 2; ++t)
        #pragma unroll
        for (int mi = 0; mi < 2; ++mi)
          aP[cb ^ 1][t][mi] = ld_afrag(h_in + (size_t)(mb + mi * 16 + lm) * HH +
                                       (2 * p + 2 + t) * 32 + lh * 8);
    }

    #pragma unroll
    for (int t = 0; t < 2; ++t) {
      bf16x16 b[4];
      #pragma unroll
      for (int g = 0; g < 4; ++g)
        b[g] = ld_bfrag(&sB[cb][t][g * 512 + lane * 16]);
      #pragma unroll
      for (int mi = 0; mi < 2; ++mi)
        #pragma unroll
        for (int g = 0; g < 4; ++g)
          acc[mi][g] = wmma_bf16(aP[cb][t][mi], b[g], acc[mi][g]);
    }

    async_stage_wait();
    __syncthreads();
  }

  // ---- fused gate epilogue (all operands in registers/LDS) ----
  #pragma unroll
  for (int mi = 0; mi < 2; ++mi) {
    #pragma unroll
    for (int r = 0; r < 8; ++r) {
      const int mloc = wave * 32 + mi * 16 + lh * 8 + r;   // 0..127 in block
      const int m    = mrow0 + mloc;
      const int tok  = sTok[mloc];
      const float* pr = sProj + tok * 64 + lm;
      float gi = acc[mi][0][r] + pr[0];     // i gate
      float gf = acc[mi][1][r] + pr[16];    // f gate
      float gg = acc[mi][2][r] + pr[32];    // g gate
      float go = acc[mi][3][r] + pr[48];    // o gate
      const int idx = m * HH + nh + lm;
      float cn = sigm(gf) * cold[mi][r] + sigm(gi) * tanhf(gg);
      float hn = sigm(go) * tanhf(cn);
      c_io[idx]  = cn;
      h_out[idx] = f2bf(hn);
    }
  }
}

// -------------------------------------------------------------------------
// Generic D[M,Npad](f32) = A[M,K](bf16) @ Bpacked + bias[Nreal].
// Wave: 2 M-tiles x 2 N-tiles. Block = 128. Grid = (Npad/32, M/128).
// -------------------------------------------------------------------------
__global__ __launch_bounds__(128) void wmma_gemm_bias_kernel(
    const __bf16* __restrict__ A, const __bf16* __restrict__ Bp,
    const float* __restrict__ bias, float* __restrict__ D,
    int M, int K, int Npad, int Nreal)
{
  const int lane = threadIdx.x & 31;
  const int wave = threadIdx.x >> 5;
  const int lm = lane & 15, lh = lane >> 4;
  const int nb = blockIdx.x * 32;
  const int mb = blockIdx.y * 128 + wave * 32;
  const int KT = K >> 5;

  f32x8 acc[2][2] = {};

  #pragma unroll 1
  for (int kt = 0; kt < KT; ++kt) {
    const int kb = kt << 5;
    bf16x16 a[2];
    #pragma unroll
    for (int mi = 0; mi < 2; ++mi)
      a[mi] = ld_afrag(A + (size_t)(mb + mi * 16 + lm) * K + kb + lh * 8);
    bf16x16 b[2];
    #pragma unroll
    for (int j = 0; j < 2; ++j) {
      const int nt = (nb >> 4) + j;
      const __bf16* bp = Bp + (((size_t)nt * KT + kt) * 32 + lane) * 16;
      b[j] = ld_bfrag(bp);
      __builtin_prefetch(bp + 512, 0, 1);   // unconditional: speculative, branch-free
    }
    #pragma unroll
    for (int mi = 0; mi < 2; ++mi)
      #pragma unroll
      for (int j = 0; j < 2; ++j)
        acc[mi][j] = wmma_bf16(a[mi], b[j], acc[mi][j]);
  }

  #pragma unroll
  for (int mi = 0; mi < 2; ++mi)
    #pragma unroll
    for (int j = 0; j < 2; ++j)
      #pragma unroll
      for (int r = 0; r < 8; ++r) {
        const int m = mb + mi * 16 + lh * 8 + r;
        const int n = nb + j * 16 + lm;
        float v = (n < Nreal) ? (acc[mi][j][r] + (bias ? bias[n] : 0.f)) : 0.f;
        D[(size_t)m * Npad + n] = v;
      }
}

// log_softmax over V=28 (logits stored with row stride 32)
__global__ void logsoftmax_kernel(const float* __restrict__ logits,
                                  float* __restrict__ out) {
  int r = blockIdx.x * blockDim.x + threadIdx.x;
  if (r >= TT * BBATCH) return;
  const float* row = logits + (size_t)r * 32;
  float m = row[0];
  for (int j = 1; j < VV; ++j) m = fmaxf(m, row[j]);
  float s = 0.f;
  for (int j = 0; j < VV; ++j) s += __expf(row[j] - m);
  float lse = m + __logf(s);
  float* o = out + (size_t)r * VV;
  for (int j = 0; j < VV; ++j) o[j] = row[j] - lse;
}

extern "C" void kernel_launch(void* const* d_in, const int* in_sizes, int n_in,
                              void* d_out, int out_size, void* d_ws, size_t ws_size,
                              hipStream_t stream) {
  (void)in_sizes; (void)n_in; (void)out_size; (void)ws_size;
  const int*   x       = (const int*)  d_in[0];
  const int*   target  = (const int*)  d_in[1];
  const float* hidden  = (const float*)d_in[2];
  const float* eps     = (const float*)d_in[3];
  const float* enc_emb = (const float*)d_in[4];
  const float* enc_Wi  = (const float*)d_in[5];
  const float* enc_Wh  = (const float*)d_in[6];
  const float* enc_b   = (const float*)d_in[7];
  const float* Wmu     = (const float*)d_in[8];
  const float* bmu     = (const float*)d_in[9];
  const float* Wlv     = (const float*)d_in[10];
  const float* blv     = (const float*)d_in[11];
  const float* dec_emb = (const float*)d_in[12];
  const float* dec_Wi  = (const float*)d_in[13];
  const float* dec_Wh  = (const float*)d_in[14];
  const float* dec_b   = (const float*)d_in[15];
  const float* Wout    = (const float*)d_in[16];
  const float* bout    = (const float*)d_in[17];

  float* out_res = (float*)d_out;                       // [T*B*V]
  float* out_mu  = out_res + (size_t)TT * BBATCH * VV;  // [B*H]
  float* out_lv  = out_mu + (size_t)BBATCH * HH;        // [B*H]

  char* ws = (char*)d_ws;
  size_t off = 0;
  auto alloc = [&](size_t bytes) -> void* {
    void* p = ws + off;
    off = (off + bytes + 255) & ~(size_t)255;
    return p;
  };
  const size_t BH = (size_t)BBATCH * HH;
  __bf16* encWhp   = (__bf16*)alloc((size_t)HH * G4H * 2);
  __bf16* decWhp   = (__bf16*)alloc((size_t)HH * G4H * 2);
  __bf16* Wmup     = (__bf16*)alloc((size_t)HH * HH * 2);
  __bf16* Wlvp     = (__bf16*)alloc((size_t)HH * HH * 2);
  __bf16* Woutp    = (__bf16*)alloc((size_t)HH * 32 * 2);
  float*  enc_proj = (float*) alloc((size_t)VV * G4H * 4);
  float*  dec_proj = (float*) alloc((size_t)VV * G4H * 4);
  __bf16* henc     = (__bf16*)alloc(2 * BH * 2);              // ping-pong
  float*  c_buf    = (float*) alloc(BH * 4);
  __bf16* hs       = (__bf16*)alloc((size_t)(TT + 1) * BH * 2); // hs[0] = z
  float*  logits   = (float*) alloc((size_t)TT * BBATCH * 32 * 4);

  // ---- weight packing ----
  {
    int tot = G4H * HH;
    pack_b_kernel<<<(tot + 255) / 256, 256, 0, stream>>>(enc_Wh, encWhp, HH, G4H, G4H);
    pack_b_kernel<<<(tot + 255) / 256, 256, 0, stream>>>(dec_Wh, decWhp, HH, G4H, G4H);
    int tot2 = HH * HH;
    pack_b_kernel<<<(tot2 + 255) / 256, 256, 0, stream>>>(Wmu, Wmup, HH, HH, HH);
    pack_b_kernel<<<(tot2 + 255) / 256, 256, 0, stream>>>(Wlv, Wlvp, HH, HH, HH);
    int tot3 = 32 * HH;
    pack_b_kernel<<<(tot3 + 255) / 256, 256, 0, stream>>>(Wout, Woutp, HH, VV, 32);
  }
  // ---- proj tables: embed @ Wi + b (folds input projection into a gather)
  {
    int tot = VV * G4H;
    embed_proj_kernel<<<(tot + 255) / 256, 256, 0, stream>>>(enc_emb, enc_Wi, enc_b, enc_proj);
    embed_proj_kernel<<<(tot + 255) / 256, 256, 0, stream>>>(dec_emb, dec_Wi, dec_b, dec_proj);
  }
  // ---- encoder ----
  init_enc_kernel<<<(int)((BH + 255) / 256), 256, 0, stream>>>(hidden, henc, c_buf);
  dim3 stepGrid(HH / 16, BBATCH / 128);  // (32, 2)
  for (int t = 0; t < TT; ++t) {
    const __bf16* hi = henc + (size_t)(t & 1) * BH;
    __bf16*       ho = henc + (size_t)((t + 1) & 1) * BH;
    lstm_step_kernel<<<stepGrid, 128, 0, stream>>>(hi, ho, c_buf, encWhp, enc_proj,
                                                   x + (size_t)t * BBATCH);
  }
  // final h at parity (TT & 1) == 0 -> henc
  // ---- mu / logvar ----
  dim3 gMu(HH / 32, BBATCH / 128);  // (16, 2)
  wmma_gemm_bias_kernel<<<gMu, 128, 0, stream>>>(henc, Wmup, bmu, out_mu, BBATCH, HH, HH, HH);
  wmma_gemm_bias_kernel<<<gMu, 128, 0, stream>>>(henc, Wlvp, blv, out_lv, BBATCH, HH, HH, HH);
  // ---- reparameterize: z -> hs[0] (bf16) and c0 (f32) ----
  reparam_kernel<<<(int)((BH + 255) / 256), 256, 0, stream>>>(out_mu, out_lv, eps, hs, c_buf);
  // ---- decoder (teacher forcing: SOS then target[:-1]) ----
  for (int t = 0; t < TT; ++t) {
    const int* tok = (t == 0) ? (const int*)nullptr : (target + (size_t)(t - 1) * BBATCH);
    lstm_step_kernel<<<stepGrid, 128, 0, stream>>>(hs + (size_t)t * BH,
                                                   hs + (size_t)(t + 1) * BH,
                                                   c_buf, decWhp, dec_proj, tok);
  }
  // ---- logits = hs[1..T] @ Wout + bout, then log_softmax ----
  dim3 gLog(1, (TT * BBATCH) / 128);  // (1, 512)
  wmma_gemm_bias_kernel<<<gLog, 128, 0, stream>>>(hs + BH, Woutp, bout, logits,
                                                  TT * BBATCH, HH, 32, VV);
  int rows = TT * BBATCH;
  logsoftmax_kernel<<<(rows + 255) / 256, 256, 0, stream>>>(logits, out_res);
}
